// simple_attention_27058293965306
// MI455X (gfx1250) — matrix operand order repaired and checked by reading, not compiled, not run on hardware
//
#include <hip/hip_runtime.h>
#include <hip/hip_bf16.h>

// ---------------------------------------------------------------------------
// Types for CDNA5 WMMA (wave32): bf16 16x16x32, f32 accumulate
// ---------------------------------------------------------------------------
typedef __attribute__((ext_vector_type(16))) __bf16 v16bf;
typedef __attribute__((ext_vector_type(8)))  __bf16 v8bf;
typedef __attribute__((ext_vector_type(8)))  float  v8f;

union Frag16 { v16bf v; v8bf h[2]; };

// Problem constants
#define BATCH 4
#define SEQ   2048
#define DMODEL 1024
#define DV 64
#define KSTEP 32
#define KSTEPS (DMODEL / KSTEP)

// LDS tile layout (elements of unsigned short), row stride padded 32->40 to
// spread ds_load_b128 rows across banks (20-DWORD stride => 16 distinct banks
// for 16 consecutive rows).
#define ROWP 40
#define A_HI_OFF 0                      // 64 rows
#define A_LO_OFF (64 * ROWP)            // 2560
#define B_HI_OFF (2 * 64 * ROWP)        // 5120, 128 rows
#define B_LO_OFF (B_HI_OFF + 128 * ROWP)// 10240
#define BUF_ELEMS (B_LO_OFF + 128 * ROWP) // 15360 elems = 30720 B per buffer

// ---------------------------------------------------------------------------
// bf16 split helpers (round-to-nearest-even; no reliance on __bf16 arithmetic)
// ---------------------------------------------------------------------------
__device__ inline unsigned short f2bf_bits(float f) {
  unsigned u = __float_as_uint(f);
  u += 0x7FFFu + ((u >> 16) & 1u);
  return (unsigned short)(u >> 16);
}
__device__ inline float bf_bits2f(unsigned short s) {
  return __uint_as_float(((unsigned)s) << 16);
}
__device__ inline void split_bf(float f, unsigned short& hi, unsigned short& lo) {
  hi = f2bf_bits(f);
  lo = f2bf_bits(f - bf_bits2f(hi));
}

// ---------------------------------------------------------------------------
// Block reductions (wave32-aware: 256 threads = 8 waves)
// ---------------------------------------------------------------------------
__device__ inline void block_reduce_sum2(float& a, float& b) {
  __shared__ float sa[8], sb[8];
  for (int o = 16; o > 0; o >>= 1) {
    a += __shfl_down(a, o, 32);
    b += __shfl_down(b, o, 32);
  }
  const int lane = threadIdx.x & 31, w = threadIdx.x >> 5;
  if (lane == 0) { sa[w] = a; sb[w] = b; }
  __syncthreads();
  if (threadIdx.x < 32) {
    a = (lane < 8) ? sa[lane] : 0.0f;
    b = (lane < 8) ? sb[lane] : 0.0f;
    for (int o = 4; o > 0; o >>= 1) {
      a += __shfl_down(a, o, 32);
      b += __shfl_down(b, o, 32);
    }
    if (lane == 0) { sa[0] = a; sb[0] = b; }
  }
  __syncthreads();
  a = sa[0]; b = sb[0];
}

__device__ inline float block_reduce_sum1(float a) {
  __shared__ float sa[8];
  for (int o = 16; o > 0; o >>= 1) a += __shfl_down(a, o, 32);
  const int lane = threadIdx.x & 31, w = threadIdx.x >> 5;
  if (lane == 0) sa[w] = a;
  __syncthreads();
  if (threadIdx.x < 32) {
    a = (lane < 8) ? sa[lane] : 0.0f;
    for (int o = 4; o > 0; o >>= 1) a += __shfl_down(a, o, 32);
    if (lane == 0) sa[0] = a;
  }
  __syncthreads();
  return sa[0];
}

// ---------------------------------------------------------------------------
// Kernel 1: split k into bf16 hi/lo (4 elems/thread, float4 loads)
// ---------------------------------------------------------------------------
__global__ __launch_bounds__(256)
void k_split_kernel(const float* __restrict__ kin,
                    unsigned short* __restrict__ khi,
                    unsigned short* __restrict__ klo) {
  const size_t i = ((size_t)blockIdx.x * 256 + threadIdx.x) * 4;
  const float4 x = *reinterpret_cast<const float4*>(kin + i);
  unsigned short h0, l0, h1, l1, h2, l2, h3, l3;
  split_bf(x.x, h0, l0); split_bf(x.y, h1, l1);
  split_bf(x.z, h2, l2); split_bf(x.w, h3, l3);
  khi[i + 0] = h0; khi[i + 1] = h1; khi[i + 2] = h2; khi[i + 3] = h3;
  klo[i + 0] = l0; klo[i + 1] = l1; klo[i + 2] = l2; klo[i + 3] = l3;
}

// ---------------------------------------------------------------------------
// Kernel 2: fused LayerNorm(q) * (1/sqrt(64)) -> bf16 hi/lo split
// ---------------------------------------------------------------------------
__global__ __launch_bounds__(256)
void q_ln_split_kernel(const float* __restrict__ q,
                       const float* __restrict__ ln_g,
                       const float* __restrict__ ln_b,
                       unsigned short* __restrict__ qhi,
                       unsigned short* __restrict__ qlo) {
  const size_t base = (size_t)blockIdx.x * DMODEL;
  const int d0 = threadIdx.x * 4;
  const float4 x = *reinterpret_cast<const float4*>(q + base + d0);
  float s  = x.x + x.y + x.z + x.w;
  float ss = x.x * x.x + x.y * x.y + x.z * x.z + x.w * x.w;
  block_reduce_sum2(s, ss);
  const float mu  = s * (1.0f / DMODEL);
  const float var = ss * (1.0f / DMODEL) - mu * mu;
  const float rs  = rsqrtf(var + 1e-6f);
  const float4 g = *reinterpret_cast<const float4*>(ln_g + d0);
  const float4 b = *reinterpret_cast<const float4*>(ln_b + d0);
  float y[4];
  y[0] = ((x.x - mu) * rs * g.x + b.x) * 0.125f;  // 1/sqrt(D_K=64)
  y[1] = ((x.y - mu) * rs * g.y + b.y) * 0.125f;
  y[2] = ((x.z - mu) * rs * g.z + b.z) * 0.125f;
  y[3] = ((x.w - mu) * rs * g.w + b.w) * 0.125f;
#pragma unroll
  for (int i = 0; i < 4; ++i) {
    unsigned short hi, lo;
    split_bf(y[i], hi, lo);
    qhi[base + d0 + i] = hi;
    qlo[base + d0 + i] = lo;
  }
}

// ---------------------------------------------------------------------------
// Kernel 3: w[d] = dot(fc_w[d,:], V[0,:])  (rank-1 collapse of the epilogue)
// ---------------------------------------------------------------------------
__global__ __launch_bounds__(256)
void w_kernel(const float* __restrict__ fcw,
              const float* __restrict__ V,
              float* __restrict__ w) {
  const int d = blockIdx.x * 256 + threadIdx.x;
  if (d < DMODEL) {
    float s = 0.0f;
#pragma unroll 8
    for (int j = 0; j < DV; ++j) s += fcw[d * DV + j] * V[j];
    w[d] = s;
  }
}

// ---------------------------------------------------------------------------
// Async stage of one k-step block tile into LDS:
// A: 64 rows x 32k (hi+lo planes), B: 128 rows x 32k (hi+lo planes).
// 256 threads issue 6 x global_load_async_to_lds_b128 each (ASYNCcnt += 6/wave).
// ---------------------------------------------------------------------------
__device__ __forceinline__ void stage_tiles(const unsigned short* __restrict__ qhi,
                                            const unsigned short* __restrict__ qlo,
                                            const unsigned short* __restrict__ khi,
                                            const unsigned short* __restrict__ klo,
                                            unsigned short* sbuf,
                                            size_t qrow0, size_t krow0,
                                            int k0, int t) {
  const unsigned sbase = (unsigned)(unsigned long long)sbuf;  // LDS byte offset
  // A planes: 64 rows * 4 chunks = 256 chunks -> 1 per thread
  {
    const int row = t >> 2, sub = t & 3;
    const unsigned lds = sbase + (unsigned)((row * ROWP + sub * 8) * 2);
    const unsigned short* gh = qhi + (qrow0 + row) * DMODEL + k0 + sub * 8;
    const unsigned short* gl = qlo + (qrow0 + row) * DMODEL + k0 + sub * 8;
    asm volatile("global_load_async_to_lds_b128 %0, %1, off"
                 :: "v"(lds), "v"(gh) : "memory");
    asm volatile("global_load_async_to_lds_b128 %0, %1, off"
                 :: "v"(lds + (unsigned)(A_LO_OFF * 2)), "v"(gl) : "memory");
  }
  // B planes: 128 rows * 4 chunks = 512 chunks -> 2 per thread
#pragma unroll
  for (int j = 0; j < 2; ++j) {
    const int c = t + j * 256;
    const int row = c >> 2, sub = c & 3;
    const unsigned lds = sbase + (unsigned)((B_HI_OFF + row * ROWP + sub * 8) * 2);
    const unsigned short* gh = khi + (krow0 + row) * DMODEL + k0 + sub * 8;
    const unsigned short* gl = klo + (krow0 + row) * DMODEL + k0 + sub * 8;
    asm volatile("global_load_async_to_lds_b128 %0, %1, off"
                 :: "v"(lds), "v"(gh) : "memory");
    asm volatile("global_load_async_to_lds_b128 %0, %1, off"
                 :: "v"(lds + (unsigned)((B_LO_OFF - B_HI_OFF) * 2)), "v"(gl) : "memory");
  }
}

// ---------------------------------------------------------------------------
// Kernel 4: masked attention GEMM via V_WMMA_F32_16X16X32_BF16.
//
// Block tile 64(M) x 128(N); 8 wave32 arranged 2(M) x 4(N), each wave a 32x32
// macro-tile (four 16x16 f32 accumulators). Per k-step (K=32):
//   - next block tile staged global->LDS via 6 async b128/thread (double buffer)
//   - s_wait_asynccnt 6 + barrier publishes the current buffer
//   - each wave: 16 x ds_load_b128 fragments -> 12 WMMAs (hi/lo split:
//     a*b ~= ah*bh + ah*bl + al*bh, f32 accumulate -> ~fp32 accuracy)
// LDS staging removes the 4x (A) / 2x (B) cross-wave duplicate L2 reads of the
// register-only version; q/k bf16 planes (67 MB) stay L2-resident (192 MB).
//
// Fragment layouts (as loaded here: contiguous k = 16h+s for A and B, wave32), h = lane>>4, l16 = lane&15:
//   A (16x32 bf16): lane=row M; elems {h*16..h*16+15}.
//   B (32x16 bf16): lane=column N; elems {h*16..h*16+15} (B = k^T, row-major k).
//   C/D (16x16 f32): VGPR r -> M = r + 8*h, N = l16.
// ---------------------------------------------------------------------------
__global__ __launch_bounds__(256)
void attn_gemm_kernel(const unsigned short* __restrict__ qhi,
                      const unsigned short* __restrict__ qlo,
                      const unsigned short* __restrict__ khi,
                      const unsigned short* __restrict__ klo,
                      const int* __restrict__ mask,
                      float* __restrict__ attn) {
  __shared__ alignas(16) unsigned short smem[2][BUF_ELEMS];  // 2 x 30 KB

  const int b    = blockIdx.z;
  const int t    = threadIdx.x;
  const int wave = t >> 5;
  const int lane = t & 31;
  const int h    = lane >> 4;
  const int l16  = lane & 15;

  const int m_block = blockIdx.x * 64;
  const int n_block = blockIdx.y * 128;
  const size_t qrow0 = (size_t)b * SEQ + m_block;   // first A row of block
  const size_t krow0 = (size_t)b * SEQ + n_block;   // first B row of block

  // This wave's sub-tiles inside the block tile
  const int wm = (wave >> 2) * 32;   // 0 or 32
  const int wn = (wave & 3) * 32;    // 0,32,64,96

  v8f c00 = {}, c01 = {}, c10 = {}, c11 = {};

  // Prologue: stage k-step 0 into buffer 0
  stage_tiles(qhi, qlo, khi, klo, smem[0], qrow0, krow0, 0, t);

  for (int i = 0; i < KSTEPS; ++i) {
    const int sel = i & 1;
    if (i + 1 < KSTEPS) {
      // Overlap: issue next buffer's DMA before waiting on current buffer.
      stage_tiles(qhi, qlo, khi, klo, smem[sel ^ 1], qrow0, krow0,
                  (i + 1) * KSTEP, t);
      asm volatile("s_wait_asynccnt 0x6" ::: "memory");  // current 6 done (in-order)
    } else {
      asm volatile("s_wait_asynccnt 0x0" ::: "memory");
    }
    __syncthreads();  // current buffer visible to all waves

    const unsigned short* s   = smem[sel];
    const unsigned short* sAh = s + A_HI_OFF;
    const unsigned short* sAl = s + A_LO_OFF;
    const unsigned short* sBh = s + B_HI_OFF;
    const unsigned short* sBl = s + B_LO_OFF;

    const int a0 = (wm + l16) * ROWP + h * 16;
    const int a1 = a0 + 16 * ROWP;
    const int b0 = (wn + l16) * ROWP + h * 16;
    const int b1 = b0 + 16 * ROWP;

    Frag16 ah0, al0, ah1, al1, bh0, bl0, bh1, bl1;
    ah0.h[0] = *reinterpret_cast<const v8bf*>(sAh + a0);
    ah0.h[1] = *reinterpret_cast<const v8bf*>(sAh + a0 + 8);
    al0.h[0] = *reinterpret_cast<const v8bf*>(sAl + a0);
    al0.h[1] = *reinterpret_cast<const v8bf*>(sAl + a0 + 8);
    ah1.h[0] = *reinterpret_cast<const v8bf*>(sAh + a1);
    ah1.h[1] = *reinterpret_cast<const v8bf*>(sAh + a1 + 8);
    al1.h[0] = *reinterpret_cast<const v8bf*>(sAl + a1);
    al1.h[1] = *reinterpret_cast<const v8bf*>(sAl + a1 + 8);
    bh0.h[0] = *reinterpret_cast<const v8bf*>(sBh + b0);
    bh0.h[1] = *reinterpret_cast<const v8bf*>(sBh + b0 + 8);
    bl0.h[0] = *reinterpret_cast<const v8bf*>(sBl + b0);
    bl0.h[1] = *reinterpret_cast<const v8bf*>(sBl + b0 + 8);
    bh1.h[0] = *reinterpret_cast<const v8bf*>(sBh + b1);
    bh1.h[1] = *reinterpret_cast<const v8bf*>(sBh + b1 + 8);
    bl1.h[0] = *reinterpret_cast<const v8bf*>(sBl + b1);
    bl1.h[1] = *reinterpret_cast<const v8bf*>(sBl + b1 + 8);

    // Four independent accumulator chains; each fragment reused 3x.
    c00 = __builtin_amdgcn_wmma_f32_16x16x32_bf16(false, ah0.v, false, bh0.v,
                                                  (short)0, c00, false, false);
    c01 = __builtin_amdgcn_wmma_f32_16x16x32_bf16(false, ah0.v, false, bh1.v,
                                                  (short)0, c01, false, false);
    c10 = __builtin_amdgcn_wmma_f32_16x16x32_bf16(false, ah1.v, false, bh0.v,
                                                  (short)0, c10, false, false);
    c11 = __builtin_amdgcn_wmma_f32_16x16x32_bf16(false, ah1.v, false, bh1.v,
                                                  (short)0, c11, false, false);
    c00 = __builtin_amdgcn_wmma_f32_16x16x32_bf16(false, ah0.v, false, bl0.v,
                                                  (short)0, c00, false, false);
    c01 = __builtin_amdgcn_wmma_f32_16x16x32_bf16(false, ah0.v, false, bl1.v,
                                                  (short)0, c01, false, false);
    c10 = __builtin_amdgcn_wmma_f32_16x16x32_bf16(false, ah1.v, false, bl0.v,
                                                  (short)0, c10, false, false);
    c11 = __builtin_amdgcn_wmma_f32_16x16x32_bf16(false, ah1.v, false, bl1.v,
                                                  (short)0, c11, false, false);
    c00 = __builtin_amdgcn_wmma_f32_16x16x32_bf16(false, al0.v, false, bh0.v,
                                                  (short)0, c00, false, false);
    c01 = __builtin_amdgcn_wmma_f32_16x16x32_bf16(false, al0.v, false, bh1.v,
                                                  (short)0, c01, false, false);
    c10 = __builtin_amdgcn_wmma_f32_16x16x32_bf16(false, al1.v, false, bh0.v,
                                                  (short)0, c10, false, false);
    c11 = __builtin_amdgcn_wmma_f32_16x16x32_bf16(false, al1.v, false, bh1.v,
                                                  (short)0, c11, false, false);

    __syncthreads();  // all waves done reading buffer sel before it is restaged
  }

  // Epilogue: mask + store four 16x16 tiles.
  const int m_base = m_block + wm;
  const int n_base = n_block + wn;
  const v8f* accs[4] = { &c00, &c01, &c10, &c11 };
#pragma unroll
  for (int tt = 0; tt < 4; ++tt) {
    const int m0 = m_base + (tt >> 1) * 16;
    const int n  = n_base + (tt & 1) * 16 + l16;
    const v8f& c = *accs[tt];
#pragma unroll
    for (int r = 0; r < 8; ++r) {
      const int m = m0 + r + h * 8;
      const size_t idx = ((size_t)b * SEQ + m) * SEQ + n;
      attn[idx] = mask[idx] ? 0.0f : c[r];
    }
  }
}

// ---------------------------------------------------------------------------
// Kernel 5: rowsum(attn) rank-1 epilogue + bias + residual.
// out[b,s,d] = rowsum[b,s] * w[d] + fc_b[d] + q[b,s,d]
// ---------------------------------------------------------------------------
__global__ __launch_bounds__(256)
void final_kernel(const float* __restrict__ attn,
                  const float* __restrict__ q,
                  const float* __restrict__ w,
                  const float* __restrict__ fcb,
                  float* __restrict__ out) {
  const size_t row = blockIdx.x;  // b*SEQ + s
  const float* arow = attn + row * SEQ;
  float s = 0.0f;
  for (int i = threadIdx.x; i < SEQ; i += 256) s += arow[i];
  const float rowsum = block_reduce_sum1(s);
  const size_t base = row * DMODEL;
  for (int d = threadIdx.x; d < DMODEL; d += 256)
    out[base + d] = rowsum * w[d] + fcb[d] + q[base + d];
}

// ---------------------------------------------------------------------------
// Launch
// ---------------------------------------------------------------------------
extern "C" void kernel_launch(void* const* d_in, const int* in_sizes, int n_in,
                              void* d_out, int out_size, void* d_ws, size_t ws_size,
                              hipStream_t stream) {
  // Inputs (setup_inputs order): q, k, v, mask, V, fc_w, fc_b, ln_g, ln_b
  const float* q    = (const float*)d_in[0];
  const float* k    = (const float*)d_in[1];
  // v = d_in[2] is unused by the reference math
  const int*   mask = (const int*)d_in[3];
  const float* V    = (const float*)d_in[4];
  const float* fcw  = (const float*)d_in[5];
  const float* fcb  = (const float*)d_in[6];
  const float* lng  = (const float*)d_in[7];
  const float* lnb  = (const float*)d_in[8];

  // Outputs: output [4,2048,1024] then attn [4,2048,2048], concatenated flat.
  float* out  = (float*)d_out;
  float* attn = out + (size_t)BATCH * SEQ * DMODEL;

  // Workspace layout (bytes): 4 x 16 MiB bf16 planes + w[1024]
  const size_t plane = (size_t)BATCH * SEQ * DMODEL * sizeof(unsigned short);
  char* ws = (char*)d_ws;
  unsigned short* qhi = (unsigned short*)(ws + 0 * plane);
  unsigned short* qlo = (unsigned short*)(ws + 1 * plane);
  unsigned short* khi = (unsigned short*)(ws + 2 * plane);
  unsigned short* klo = (unsigned short*)(ws + 3 * plane);
  float* w = (float*)(ws + 4 * plane);

  const int rows = BATCH * SEQ;  // 8192

  // 1) bf16 hi/lo split of k (8M elems, 4/thread)
  k_split_kernel<<<rows * DMODEL / (256 * 4), 256, 0, stream>>>(k, khi, klo);

  // 2) LayerNorm(q)/sqrt(64) -> bf16 hi/lo
  q_ln_split_kernel<<<rows, 256, 0, stream>>>(q, lng, lnb, qhi, qlo);

  // 3) w[d] = fc_w[d,:] . V
  w_kernel<<<(DMODEL + 255) / 256, 256, 0, stream>>>(fcw, V, w);

  // 4) masked WMMA GEMM with async-LDS double-buffered staging:
  //    block tile 64x128, grid (2048/64, 2048/128, B)
  attn_gemm_kernel<<<dim3(SEQ / 64, SEQ / 128, BATCH), 256, 0, stream>>>(
      qhi, qlo, khi, klo, mask, attn);

  // 5) rowsum + rank-1 epilogue + residual
  final_kernel<<<rows, 256, 0, stream>>>(attn, q, w, fcb, out);
}
